// HyperbolicCausalSelfAttention_31464930410813
// MI455X (gfx1250) — matrix operand order repaired and checked by reading, not compiled, not run on hardware
//
#include <hip/hip_runtime.h>
#include <hip/hip_bf16.h>
#include <cstdint>
#include <cstddef>

typedef __bf16 bf16;
typedef __attribute__((ext_vector_type(8)))  bf16  bf16x8;
typedef __attribute__((ext_vector_type(16))) bf16  bf16x16;
typedef __attribute__((ext_vector_type(8)))  float v8f;

#define WMMA_BF16(a, b, c) \
  __builtin_amdgcn_wmma_f32_16x16x32_bf16(false, (a), false, (b), (short)0, (c), false, false)

constexpr int BB   = 2;
constexpr int T    = 2048;
constexpr int DIM  = 1040;
constexpr int H    = 16;
constexpr int HD   = 64;          // spatial head dim
constexpr int HDE  = HD + 1;      // 65 = lorentz + spatial
constexpr int KP   = 1056;        // K dim padded to multiple of 32
constexpr int DIMP = 1056;        // N dim padded (66 tiles of 16)
constexpr int M    = BB * T;      // 4096 rows
constexpr int QKD  = 96;          // padded q/k head dim (65 -> 96 = 3*32)
constexpr int VD   = 80;          // padded v head dim (65 -> 80 = 5*16)
// GEMM blocking: wave computes 64 (M) x 32 (N)
constexpr int MB   = (M / 16) / 4;     // 64 m-blocks
constexpr int NB   = (DIMP / 16) / 2;  // 33 n-blocks

// ---------------- fragment loaders ----------------
// A-matrix (16x32 bf16): lane m holds K 0..15; lane m+16 holds K 16..31.
__device__ inline bf16x16 load_a(const bf16* p) {
  bf16x8 lo = *reinterpret_cast<const bf16x8*>(p);
  bf16x8 hi = *reinterpret_cast<const bf16x8*>(p + 8);
  bf16x16 r;
#pragma unroll
  for (int e = 0; e < 8; ++e) { r[e] = lo[e]; r[e + 8] = hi[e]; }
  return r;
}
// B-matrix (32x16 bf16): lane n = column n; lanes 0-15 hold K 0..15, lanes 16-31 hold K 16..31.
__device__ inline bf16x16 load_b(const bf16* p) {
  bf16x8 lo = *reinterpret_cast<const bf16x8*>(p);
  bf16x8 hi = *reinterpret_cast<const bf16x8*>(p + 8);
  bf16x16 r;
#pragma unroll
  for (int e = 0; e < 8; ++e) { r[e] = lo[e]; r[e + 8] = hi[e]; }
  return r;
}

// ---------------- kernels ----------------

// f32 (rows x cols) -> bf16 (rowsp x colsp), zero padded rows/cols
__global__ void k_cvt(const float* __restrict__ src, bf16* __restrict__ dst,
                      int rows, int cols, int rowsp, int colsp) {
  int tid = blockIdx.x * blockDim.x + threadIdx.x;
  int total = rowsp * colsp;
  if (tid >= total) return;
  int r = tid / colsp, c = tid - r * colsp;
  dst[tid] = (r < rows && c < cols) ? (bf16)src[(size_t)r * cols + c] : (bf16)0.0f;
}

__global__ void k_zero_bf16(bf16* __restrict__ dst, int n) {
  int tid = blockIdx.x * blockDim.x + threadIdx.x;
  if (tid < n) dst[tid] = (bf16)0.0f;
}

// QKV projections: C = X (M x KP) * W^T.  W stored (DIMP x KP) bf16 row-major (rows >= DIM zero).
// One wave -> 64x32 output tile (4 M-tiles x 2 N-tiles), 8 WMMAs per K-step.
__global__ void k_gemm_qkv(const bf16* __restrict__ xb,
                           const bf16* __restrict__ wq, const bf16* __restrict__ wk,
                           const bf16* __restrict__ wv,
                           bf16* __restrict__ qraw, bf16* __restrict__ kraw,
                           bf16* __restrict__ vraw) {
  const int lane = threadIdx.x;
  const int mb = blockIdx.x / NB, nb = blockIdx.x - mb * NB;
  const bf16* W; bf16* out;
  if (blockIdx.y == 0)      { W = wq; out = qraw; }
  else if (blockIdx.y == 1) { W = wk; out = kraw; }
  else                      { W = wv; out = vraw; }
  const int m0 = mb * 64, n0 = nb * 32;
  const int khalf = lane >> 4, idx = lane & 15;
  const bf16* arow[4];
  const bf16* brow[2];
#pragma unroll
  for (int im = 0; im < 4; ++im) arow[im] = xb + (size_t)(m0 + im * 16 + idx) * KP;
#pragma unroll
  for (int jn = 0; jn < 2; ++jn) brow[jn] = W + (size_t)(n0 + jn * 16 + idx) * KP;
  v8f acc[4][2];
#pragma unroll
  for (int im = 0; im < 4; ++im)
#pragma unroll
    for (int jn = 0; jn < 2; ++jn) acc[im][jn] = v8f{};
  for (int k0 = 0; k0 < KP; k0 += 32) {
    bf16x16 af[4], bfr[2];
#pragma unroll
    for (int im = 0; im < 4; ++im) af[im] = load_a(arow[im] + k0 + khalf * 16);
#pragma unroll
    for (int jn = 0; jn < 2; ++jn) bfr[jn] = load_b(brow[jn] + k0 + khalf * 16);
#pragma unroll
    for (int im = 0; im < 4; ++im)
#pragma unroll
      for (int jn = 0; jn < 2; ++jn) acc[im][jn] = WMMA_BF16(af[im], bfr[jn], acc[im][jn]);
  }
  // C layout: VGPR i -> row +i+8*khalf, col +idx
#pragma unroll
  for (int im = 0; im < 4; ++im)
#pragma unroll
    for (int jn = 0; jn < 2; ++jn) {
      const int col = n0 + jn * 16 + idx;
      if (col < DIM) {
#pragma unroll
        for (int i = 0; i < 8; ++i)
          out[(size_t)(m0 + im * 16 + i + 8 * khalf) * DIM + col] = (bf16)acc[im][jn][i];
      }
    }
}

// Hyperbolic rotary: per (b,t,h) build qpad/kpad (96-wide, k0 negated) and vT (dim-major).
__global__ void k_rotary(const bf16* __restrict__ qraw, const bf16* __restrict__ kraw,
                         const bf16* __restrict__ vraw,
                         bf16* __restrict__ qpad, bf16* __restrict__ kpad,
                         bf16* __restrict__ vT) {
  int tid = blockIdx.x * blockDim.x + threadIdx.x;
  if (tid >= BB * T * H) return;
  const int h = tid & 15;
  const int t = (tid >> 4) & (T - 1);
  const int b = tid >> 15;
  const size_t rbase = ((size_t)(b * T + t)) * DIM + h * HDE;
  bf16* qp = qpad + (((size_t)(b * T + t)) * H + h) * QKD;
  bf16* kp = kpad + (((size_t)(b * T + t)) * H + h) * QKD;
  float qs = 0.f, ks = 0.f;
  const float tf = (float)t;
#pragma unroll 4
  for (int i = 0; i < HD / 2; ++i) {
    float invf = __powf(10000.0f, -(2.0f * i) / (64.0f - 2e-6f));
    float ang = tf * invf;
    float c = cosf(ang), s = sinf(ang);
    float x1 = (float)qraw[rbase + 1 + i], x2 = (float)qraw[rbase + 1 + HD / 2 + i];
    float y1 = fminf(fmaxf(x1 * c + x2 * s, -1000.f), 1000.f);
    float y2 = fminf(fmaxf(-x1 * s + x2 * c, -1000.f), 1000.f);
    qp[1 + i] = (bf16)y1; qp[1 + HD / 2 + i] = (bf16)y2;
    qs += y1 * y1 + y2 * y2;
    x1 = (float)kraw[rbase + 1 + i]; x2 = (float)kraw[rbase + 1 + HD / 2 + i];
    y1 = fminf(fmaxf(x1 * c + x2 * s, -1000.f), 1000.f);
    y2 = fminf(fmaxf(-x1 * s + x2 * c, -1000.f), 1000.f);
    kp[1 + i] = (bf16)y1; kp[1 + HD / 2 + i] = (bf16)y2;
    ks += y1 * y1 + y2 * y2;
  }
  qs = fminf(qs, 1.0e6f);
  ks = fminf(ks, 1.0e6f);
  qp[0] = (bf16)sqrtf(1.0f + qs + 1e-6f);
  kp[0] = (bf16)(-sqrtf(1.0f + ks + 1e-6f));   // fold Lorentz subtraction into dot product
#pragma unroll
  for (int d = HDE; d < QKD; ++d) { qp[d] = (bf16)0.0f; kp[d] = (bf16)0.0f; }
  bf16* vtb = vT + ((size_t)(b * H + h) * VD) * T + t;
#pragma unroll 4
  for (int d = 0; d < HDE; ++d) vtb[(size_t)d * T] = vraw[rbase + d];
#pragma unroll
  for (int d = HDE; d < VD; ++d) vtb[(size_t)d * T] = (bf16)0.0f;
}

// Flash attention: one wave per (b, h, 32-row q block = 2 q-tiles), streaming 32-key chunks.
// K and V fragments are loaded once per chunk and reused by both q-tiles.
__global__ void k_attn(const bf16* __restrict__ qpad, const bf16* __restrict__ kpad,
                       const bf16* __restrict__ vT, bf16* __restrict__ y) {
  __shared__ bf16 sP[32 * 16];   // one 16x32 P tile staged in A-matrix layout (1 KB)
  const int lane = threadIdx.x;
  const int blk  = blockIdx.x;
  const int qt2 = blk & (T / 32 - 1);
  const int h   = (blk >> 6) & (H - 1);
  const int b   = blk >> 10;
  const int q0  = qt2 * 32;
  const int khalf = lane >> 4, idx = lane & 15;

  // Q A-fragments: 2 q-tiles x 3 K-steps of the 96-dim dot product
  bf16x16 qa[2][3];
#pragma unroll
  for (int u = 0; u < 2; ++u) {
    const bf16* qrow = qpad + (((size_t)(b * T + q0 + u * 16 + idx)) * H + h) * QKD;
#pragma unroll
    for (int s = 0; s < 3; ++s) qa[u][s] = load_a(qrow + s * 32 + khalf * 16);
  }

  v8f o[2][5];
  float mrow[2][8], lrow[2][8];
#pragma unroll
  for (int u = 0; u < 2; ++u) {
#pragma unroll
    for (int j = 0; j < 5; ++j) o[u][j] = v8f{};
#pragma unroll
    for (int i = 0; i < 8; ++i) { mrow[u][i] = -1e30f; lrow[u][i] = 0.f; }
  }

  const bf16* vbase = vT + ((size_t)(b * H + h) * VD) * T;

  for (int s0 = 0; s0 < T; s0 += 32) {
    // K fragments for this 32-key chunk (shared by both q-tiles)
    bf16x16 kb[2][3];
#pragma unroll
    for (int ks = 0; ks < 3; ++ks) {
      kb[0][ks] = load_b(kpad + (((size_t)(b * T + s0 + idx)) * H + h) * QKD + ks * 32 + khalf * 16);
      kb[1][ks] = load_b(kpad + (((size_t)(b * T + s0 + 16 + idx)) * H + h) * QKD + ks * 32 + khalf * 16);
    }
    bf16x16 pa[2];
    float alpha[2][8];
#pragma unroll
    for (int u = 0; u < 2; ++u) {
      v8f S0 = {}, S1 = {};
#pragma unroll
      for (int ks = 0; ks < 3; ++ks) {
        S0 = WMMA_BF16(qa[u][ks], kb[0][ks], S0);
        S1 = WMMA_BF16(qa[u][ks], kb[1][ks], S1);
      }
      float p0[8], p1[8];
#pragma unroll
      for (int i = 0; i < 8; ++i) {
        float a0 = fminf(fmaxf(S0[i] * 0.125f, -100.f), 100.f);
        float a1 = fminf(fmaxf(S1[i] * 0.125f, -100.f), 100.f);
        float mx = fmaxf(a0, a1);
#pragma unroll
        for (int d = 1; d < 16; d <<= 1) mx = fmaxf(mx, __shfl_xor(mx, d, 32));
        float mn = fmaxf(mrow[u][i], mx);
        alpha[u][i] = __expf(mrow[u][i] - mn);
        mrow[u][i]  = mn;
        p0[i] = __expf(a0 - mn);
        p1[i] = __expf(a1 - mn);
        float r = p0[i] + p1[i];
#pragma unroll
        for (int d = 1; d < 16; d <<= 1) r += __shfl_xor(r, d, 32);
        lrow[u][i] = lrow[u][i] * alpha[u][i] + r;
      }
      // stage P (C layout) into LDS in A-matrix layout:
      // element (m,k) -> lane m + 16*(k>>4), u16 slot (k&15)
      __syncthreads();
#pragma unroll
      for (int i = 0; i < 8; ++i) {
        const int m = i + 8 * khalf;
        int k = idx;        // columns 0..15 (tile 0)
        sP[(m + (k >> 4) * 16) * 16 + (k & 15)] = (bf16)p0[i];
        k = idx + 16;       // columns 16..31 (tile 1)
        sP[(m + (k >> 4) * 16) * 16 + (k & 15)] = (bf16)p1[i];
      }
      __syncthreads();
      pa[u] = load_b(&sP[lane * 16]);   // 16 consecutive u16s = this lane's A fragment
    }
    // rescale accumulators and add P.V ; V fragments shared by both q-tiles
#pragma unroll
    for (int j = 0; j < 5; ++j) {
      bf16x16 vb = load_b(vbase + (size_t)(j * 16 + idx) * T + s0 + khalf * 16);
#pragma unroll
      for (int u = 0; u < 2; ++u) {
        v8f c;
#pragma unroll
        for (int i = 0; i < 8; ++i) c[i] = o[u][j][i] * alpha[u][i];
        o[u][j] = WMMA_BF16(pa[u], vb, c);
      }
    }
  }
  // finalize and scatter into padded y (M x KP, bf16)
#pragma unroll
  for (int u = 0; u < 2; ++u)
#pragma unroll
    for (int j = 0; j < 5; ++j)
#pragma unroll
      for (int i = 0; i < 8; ++i) {
        const int col = j * 16 + idx;
        if (col < HDE) {
          const int row = q0 + u * 16 + i + 8 * khalf;
          y[(size_t)(b * T + row) * KP + h * HDE + col] = (bf16)(o[u][j][i] / lrow[u][i]);
        }
      }
}

// Output projection: out(f32) = Y (M x KP bf16) * Wp^T, 64x32 tile per wave.
__global__ void k_gemm_out(const bf16* __restrict__ yb, const bf16* __restrict__ wpb,
                           float* __restrict__ out) {
  const int lane = threadIdx.x;
  const int mb = blockIdx.x / NB, nb = blockIdx.x - mb * NB;
  const int m0 = mb * 64, n0 = nb * 32;
  const int khalf = lane >> 4, idx = lane & 15;
  const bf16* arow[4];
  const bf16* brow[2];
#pragma unroll
  for (int im = 0; im < 4; ++im) arow[im] = yb + (size_t)(m0 + im * 16 + idx) * KP;
#pragma unroll
  for (int jn = 0; jn < 2; ++jn) brow[jn] = wpb + (size_t)(n0 + jn * 16 + idx) * KP;
  v8f acc[4][2];
#pragma unroll
  for (int im = 0; im < 4; ++im)
#pragma unroll
    for (int jn = 0; jn < 2; ++jn) acc[im][jn] = v8f{};
  for (int k0 = 0; k0 < KP; k0 += 32) {
    bf16x16 af[4], bfr[2];
#pragma unroll
    for (int im = 0; im < 4; ++im) af[im] = load_a(arow[im] + k0 + khalf * 16);
#pragma unroll
    for (int jn = 0; jn < 2; ++jn) bfr[jn] = load_b(brow[jn] + k0 + khalf * 16);
#pragma unroll
    for (int im = 0; im < 4; ++im)
#pragma unroll
      for (int jn = 0; jn < 2; ++jn) acc[im][jn] = WMMA_BF16(af[im], bfr[jn], acc[im][jn]);
  }
#pragma unroll
  for (int im = 0; im < 4; ++im)
#pragma unroll
    for (int jn = 0; jn < 2; ++jn) {
      const int col = n0 + jn * 16 + idx;
      if (col < DIM) {
#pragma unroll
        for (int i = 0; i < 8; ++i)
          out[(size_t)(m0 + im * 16 + i + 8 * khalf) * DIM + col] = acc[im][jn][i];
      }
    }
}

// ---------------- launch ----------------
constexpr size_t SZ_XB  = (size_t)M * KP * 2;
constexpr size_t SZ_W   = (size_t)DIMP * KP * 2;   // row-padded weights
constexpr size_t SZ_RAW = (size_t)M * DIM * 2;
constexpr size_t SZ_QP  = (size_t)M * H * QKD * 2;
constexpr size_t SZ_VT  = (size_t)BB * H * VD * T * 2;
constexpr size_t SZ_Y   = SZ_XB;

extern "C" void kernel_launch(void* const* d_in, const int* in_sizes, int n_in,
                              void* d_out, int out_size, void* d_ws, size_t ws_size,
                              hipStream_t stream) {
  const float* x  = (const float*)d_in[0];
  const float* Wq = (const float*)d_in[1];
  const float* Wk = (const float*)d_in[2];
  const float* Wv = (const float*)d_in[3];
  const float* Wp = (const float*)d_in[4];

  char* p = (char*)d_ws;
  bf16* xb   = (bf16*)p; p += SZ_XB;
  bf16* wqb  = (bf16*)p; p += SZ_W;
  bf16* wkb  = (bf16*)p; p += SZ_W;
  bf16* wvb  = (bf16*)p; p += SZ_W;
  bf16* wpb  = (bf16*)p; p += SZ_W;
  bf16* qraw = (bf16*)p; p += SZ_RAW;
  bf16* kraw = (bf16*)p; p += SZ_RAW;
  bf16* vraw = (bf16*)p; p += SZ_RAW;
  bf16* qpad = (bf16*)p; p += SZ_QP;
  bf16* kpad = (bf16*)p; p += SZ_QP;
  bf16* vT   = (bf16*)p; p += SZ_VT;
  bf16* yb   = (bf16*)p; p += SZ_Y;

  const int thr = 256;
  // 1. precision conversion + K/N padding
  k_cvt<<<(M * KP + thr - 1) / thr, thr, 0, stream>>>(x, xb, M, DIM, M, KP);
  k_cvt<<<(DIMP * KP + thr - 1) / thr, thr, 0, stream>>>(Wq, wqb, DIM, DIM, DIMP, KP);
  k_cvt<<<(DIMP * KP + thr - 1) / thr, thr, 0, stream>>>(Wk, wkb, DIM, DIM, DIMP, KP);
  k_cvt<<<(DIMP * KP + thr - 1) / thr, thr, 0, stream>>>(Wv, wvb, DIM, DIM, DIMP, KP);
  k_cvt<<<(DIMP * KP + thr - 1) / thr, thr, 0, stream>>>(Wp, wpb, DIM, DIM, DIMP, KP);
  // 2. QKV projections (WMMA, 64x32 per wave)
  k_gemm_qkv<<<dim3(MB * NB, 3), 32, 0, stream>>>(xb, wqb, wkb, wvb, qraw, kraw, vraw);
  // 3. hyperbolic rotary + layout transforms
  k_rotary<<<(BB * T * H + thr - 1) / thr, thr, 0, stream>>>(qraw, kraw, vraw, qpad, kpad, vT);
  // 4. zero padded Y then attention (WMMA, 32 q-rows per wave)
  k_zero_bf16<<<(M * KP + thr - 1) / thr, thr, 0, stream>>>(yb, M * KP);
  k_attn<<<BB * H * (T / 32), 32, 0, stream>>>(qpad, kpad, vT, yb);
  // 5. output projection (WMMA) -> f32 d_out
  k_gemm_out<<<MB * NB, 32, 0, stream>>>(yb, wpb, (float*)d_out);
}
